// GNN_85547158602358
// MI455X (gfx1250) — compile-verified, matrix operand order checked by reading, not run on hardware
//
#include <hip/hip_runtime.h>
#include <hip/hip_bf16.h>

// ---------------- problem constants ----------------
#define M_  4
#define DEPTH_ 8
#define L_  256
#define B_  8192
#define NODES_PAD 256          // 255 real tree nodes, padded to 256
#define TEMP_INV 14.285714285714286f   // 1/0.07
#define GAMMA_ 1.0f
#define LAMBD_ 1.0f

typedef unsigned short u16t;
typedef unsigned int   u32t;
typedef unsigned long long u64t;

typedef __attribute__((ext_vector_type(16))) __bf16 v16bf;
typedef __attribute__((ext_vector_type(8)))  float  v8f;
typedef __attribute__((ext_vector_type(8)))  unsigned int v8u;

// ---------------- helpers ----------------
__device__ inline u16t f2bf(float f) {                 // round-to-nearest-even fp32->bf16
    u32t u = __float_as_uint(f);
    u32t r = u + 0x7FFFu + ((u >> 16) & 1u);
    return (u16t)(r >> 16);
}

__device__ inline float blockReduceSum(float v, float* red) {
    int t = threadIdx.x;
    red[t] = v;
    __syncthreads();
    for (int s = blockDim.x >> 1; s > 0; s >>= 1) {
        if (t < s) red[t] += red[t + s];
        __syncthreads();
    }
    float r = red[0];
    __syncthreads();
    return r;
}

// order-preserving (value, row) packing for 64-bit atomicMin column argmin
__device__ inline u64t packMin(float f, u32t row) {
    u32t u = __float_as_uint(f);
    u32t key = (u & 0x80000000u) ? ~u : (u | 0x80000000u);
    return ((u64t)key << 32) | (u64t)row;
}
__device__ inline float unpackVal(u64t p) {
    u32t key = (u32t)(p >> 32);
    u32t u = (key & 0x80000000u) ? (key & 0x7FFFFFFFu) : ~key;
    return __uint_as_float(u);
}

// Load a 16x32 bf16 tile (row-major, stride 256 elems) into the WMMA A/B
// fragment layout: lanes 0-15 hold rows 0-15 with K={0..7,16..23},
// lanes 16-31 hold rows 0-15 with K={8..15,24..31}; each VGPR = 2 packed bf16.
__device__ inline v16bf loadTile(const u16t* base, int half, int r) {
    const u16t* row = base + r * 256;
    v8u t;
#pragma unroll
    for (int i = 0; i < 8; ++i) {
        int k = ((i < 4) ? (2 * i) : (16 + 2 * (i - 4))) + half * 8;
        t[i] = *(const u32t*)(row + k);
    }
    return __builtin_bit_cast(v16bf, t);
}

// ---------------- kernels ----------------
__global__ void initK(float* acc, int* cnt, u64t* colmin) {
    int t = threadIdx.x;                    // 1024 threads
    if (t < 16) acc[t] = 0.0f;
    cnt[t] = 0;                             // M_*NODES_PAD == 1024
    colmin[t] = ~0ull;
}

// xs -> bf16 + row squared-norms
__global__ void prepX(const float* __restrict__ xs, u16t* __restrict__ xbf,
                      float* __restrict__ x2) {
    __shared__ float red[256];
    int m = blockIdx.x >> 13, b = blockIdx.x & (B_ - 1), k = threadIdx.x;
    size_t idx = ((size_t)(m * B_ + b)) * L_ + k;
    float x = xs[idx];
    xbf[idx] = f2bf(x);
    float s = blockReduceSum(x * x, red);
    if (k == 0) x2[m * B_ + b] = s;
}

// tree_emb -> codebook bf16, normalized bf16, squared norms, KL reduction
__global__ void prepE(const float* __restrict__ tree, u16t* __restrict__ ebf,
                      u16t* __restrict__ nbf, float* __restrict__ e2, float* acc) {
    __shared__ float red[256];
    int m = blockIdx.x >> 8, node = blockIdx.x & 255, k = threadIdx.x;
    size_t oidx = ((size_t)(m * NODES_PAD + node)) * L_ + k;
    if (node >= 255) {                       // zero padding row
        ebf[oidx] = 0; nbf[oidx] = 0;
        if (k == 0) e2[m * NODES_PAD + node] = 0.0f;
        return;
    }
    const float* row = tree + ((size_t)(m * 255 + node)) * (2 * L_);
    float mu = row[k];
    float lv = row[L_ + k];
    float s2 = blockReduceSum(mu * mu, red);
    if (k == 0) e2[m * NODES_PAD + node] = s2;
    ebf[oidx] = f2bf(mu);
    float nrm = fmaxf(sqrtf(s2), 1e-12f);
    nbf[oidx] = f2bf(mu / nrm);
    float klv = 0.5f * (expf(lv) + mu * mu - 1.0f - lv);
    float ks = blockReduceSum(klv, red);
    if (k == 0) atomicAdd(&acc[0], ks);
}

// Dot[m][b][node] = X_m @ E_m^T via v_wmma_f32_16x16x32_bf16.
// 32768 16x16 tiles, 8 waves/block, K-dim 256 in 8 WMMA steps per tile.
__global__ void dotGemm(const u16t* __restrict__ X, const u16t* __restrict__ E,
                        float* __restrict__ Dot) {
    int wave = threadIdx.x >> 5, lane = threadIdx.x & 31;
    int tile = blockIdx.x * 8 + wave;
    int m = tile >> 13;                      // 8192 tiles per modality
    int t = tile & 8191;
    int trow = t >> 4, tcol = t & 15;
    int half = lane >> 4, r = lane & 15;
    const u16t* Xm = X + ((size_t)m * B_ + trow * 16) * L_;
    const u16t* Em = E + ((size_t)m * NODES_PAD + tcol * 16) * L_;
    v8f c = {};
#pragma unroll
    for (int kk = 0; kk < L_; kk += 32) {
        v16bf a = loadTile(Xm + kk, half, r);
        v16bf b = loadTile(Em + kk, half, r);   // row-major E rows == B^T columns
        c = __builtin_amdgcn_wmma_f32_16x16x32_bf16(false, a, false, b,
                                                    (short)0, c, false, false);
    }
    float* D = Dot + ((size_t)m * B_ + trow * 16) * NODES_PAD + tcol * 16 + r;
#pragma unroll
    for (int j = 0; j < 8; ++j)
        D[(size_t)(half * 8 + j) * NODES_PAD] = c[j];
}

// Gram[p][i][j] = nhat_p(i) . nhat_{p+1}(j) over all padded nodes (3 pairs)
__global__ void gramGemm(const u16t* __restrict__ N, float* __restrict__ G) {
    int wave = threadIdx.x >> 5, lane = threadIdx.x & 31;
    int tile = blockIdx.x * 8 + wave;        // 768 tiles total
    int p = tile >> 8;
    int t = tile & 255;
    int trow = t >> 4, tcol = t & 15;
    int half = lane >> 4, r = lane & 15;
    const u16t* A = N + ((size_t)p * NODES_PAD + trow * 16) * L_;
    const u16t* Bm = N + ((size_t)(p + 1) * NODES_PAD + tcol * 16) * L_;
    v8f c = {};
#pragma unroll
    for (int kk = 0; kk < L_; kk += 32) {
        v16bf a = loadTile(A + kk, half, r);
        v16bf b = loadTile(Bm + kk, half, r);
        c = __builtin_amdgcn_wmma_f32_16x16x32_bf16(false, a, false, b,
                                                    (short)0, c, false, false);
    }
    float* D = G + ((size_t)p * NODES_PAD + trow * 16) * NODES_PAD + tcol * 16 + r;
#pragma unroll
    for (int j = 0; j < 8; ++j)
        D[(size_t)(half * 8 + j) * NODES_PAD] = c[j];
}

// One level of the tree VQ: choose best child, counts, col-min, EC loss.
__global__ void levelK(const float* __restrict__ x2, const float* __restrict__ e2,
                       const float* __restrict__ Dot, int* __restrict__ idxi,
                       float* __restrict__ outIdx, int* __restrict__ cnt,
                       u64t* __restrict__ colmin, float* acc, int d) {
    __shared__ float red[256];
    int t = blockIdx.x * blockDim.x + threadIdx.x;   // M_*B_ threads
    int m = t >> 13, b = t & (B_ - 1);
    int off = (1 << d) - 1;
    const float* drow = Dot + ((size_t)(m * B_ + b)) * NODES_PAD;
    float xb2 = x2[m * B_ + b];
    int chosen; float dc;
    if (d == 0) {
        chosen = 0;
        dc = xb2 + e2[m * NODES_PAD + 0] - 2.0f * drow[0];
        atomicMin(&colmin[m * NODES_PAD + 0], packMin(dc, (u32t)b));
    } else {
        int p = idxi[(m * DEPTH_ + (d - 1)) * B_ + b];
        int n0 = off + 2 * p, n1 = n0 + 1;
        float d0 = xb2 + e2[m * NODES_PAD + n0] - 2.0f * drow[n0];
        float d1 = xb2 + e2[m * NODES_PAD + n1] - 2.0f * drow[n1];
        atomicMin(&colmin[m * NODES_PAD + n0], packMin(d0, (u32t)b));
        atomicMin(&colmin[m * NODES_PAD + n1], packMin(d1, (u32t)b));
        if (d1 < d0) { chosen = 2 * p + 1; dc = d1; }
        else         { chosen = 2 * p;     dc = d0; }
    }
    idxi[(m * DEPTH_ + d) * B_ + b] = chosen;
    outIdx[(m * DEPTH_ + d) * B_ + b] = (float)chosen;
    atomicAdd(&cnt[m * NODES_PAD + off + chosen], 1);
    // loss_ec = 2 * sum_b dist[b, idx_b] / (B*L)
    float s = blockReduceSum(2.0f * dc / (float)((size_t)B_ * L_), red);
    if (threadIdx.x == 0) atomicAdd(&acc[1], s);
}

// loss_ce: per-node column winner (or dist[0][node] if column unreachable)
__global__ void colK(const float* __restrict__ x2, const float* __restrict__ e2,
                     const float* __restrict__ Dot, const u64t* __restrict__ colmin,
                     float* acc) {
    __shared__ float red[256];
    int m = blockIdx.x, node = threadIdx.x;
    float contrib = 0.0f;
    if (node < 255) {
        int level = 31 - __clz(node + 1);
        int K = 1 << level;
        u64t v = colmin[m * NODES_PAD + node];
        float val;
        if (v == ~0ull)   // no candidate rows (masked column): jnp argmin -> row 0
            val = x2[m * B_] + e2[m * NODES_PAD + node]
                - 2.0f * Dot[((size_t)m * B_) * NODES_PAD + node];
        else
            val = unpackVal(v);
        contrib = 2.0f * val / (float)(K * L_);
    }
    float s = blockReduceSum(contrib, red);
    if (threadIdx.x == 0) atomicAdd(&acc[1], s);
}

// z_all gather: one wave per (m,d,b) row, 1 KB per row, float4 traffic.
__global__ void zGather(const float* __restrict__ tree, const int* __restrict__ idxi,
                        float* __restrict__ outZ) {
    int lane = threadIdx.x & 31;
    int row = blockIdx.x * 8 + (threadIdx.x >> 5);   // M_*DEPTH_*B_ rows
    int m = row >> 16;
    int rem = row & 65535;
    int d = rem >> 13, b = rem & (B_ - 1);
    int off = (1 << d) - 1;
    int node = off + idxi[(m * DEPTH_ + d) * B_ + b];
    const float4* src = (const float4*)(tree + ((size_t)(m * 255 + node)) * (2 * L_));
    float4* dst = (float4*)(outZ + (size_t)row * L_);
    dst[lane]      = src[lane];
    dst[lane + 32] = src[lane + 32];
}

// tree-Wasserstein between per-level histograms of adjacent modalities
__global__ void wassK(const int* __restrict__ cnt, float* acc) {
    __shared__ float p1s[128], p2s[128];
    int p = blockIdx.x >> 3, d = blockIdx.x & 7;
    int off = (1 << d) - 1, K = 1 << d;
    int t = threadIdx.x;
    if (t < K) {
        float inv = 1.0f / ((float)B_ + 1e-8f);
        p1s[t] = (float)cnt[p * NODES_PAD + off + t] * inv;
        p2s[t] = (float)cnt[(p + 1) * NODES_PAD + off + t] * inv;
    }
    __syncthreads();
    if (t == 0) {
        float s = 0.0f;
        int cur = K;
        while (cur > 1) {
            for (int i = 0; i < cur; ++i) s += fabsf(p1s[i] - p2s[i]);
            for (int i = 0; i < cur / 2; ++i) {
                p1s[i] = p1s[2 * i] + p1s[2 * i + 1];
                p2s[i] = p2s[2 * i] + p2s[2 * i + 1];
            }
            cur >>= 1;
        }
        atomicAdd(&acc[2], GAMMA_ * s);
    }
}

// per-(pair,level) logsumexp over the K codebook columns, weighted by counts
__global__ void lseK(const float* __restrict__ G, const int* __restrict__ cnt,
                     float* __restrict__ lse) {
    int p = blockIdx.x >> 3, d = blockIdx.x & 7;
    int off = (1 << d) - 1, K = 1 << d;
    int a = threadIdx.x;
    if (a < K) {
        const float* grow = G + ((size_t)p * NODES_PAD + off + a) * NODES_PAD + off;
        float s = 0.0f;
        for (int k = 0; k < K; ++k)
            s += (float)cnt[(p + 1) * NODES_PAD + off + k] * expf(grow[k] * TEMP_INV);
        lse[p * NODES_PAD + off + a] = logf(s);
    }
}

// InfoNCE: -mean_i( S[am_i,an_i]/T - lse[am_i] ) over 3 pairs x 8 depths
__global__ void nceK(const float* __restrict__ G, const float* __restrict__ lse,
                     const int* __restrict__ idxi, float* acc) {
    __shared__ float red[256];
    int id = blockIdx.x * blockDim.x + threadIdx.x;  // 3*DEPTH_*B_
    int p = id / (DEPTH_ * B_);
    int rem = id % (DEPTH_ * B_);
    int d = rem >> 13, i = rem & (B_ - 1);
    int off = (1 << d) - 1;
    int am = idxi[(p * DEPTH_ + d) * B_ + i];
    int an = idxi[((p + 1) * DEPTH_ + d) * B_ + i];
    float val = G[((size_t)p * NODES_PAD + off + am) * NODES_PAD + off + an] * TEMP_INV
              - lse[p * NODES_PAD + off + am];
    float s = blockReduceSum(val, red);
    if (threadIdx.x == 0) atomicAdd(&acc[2], -LAMBD_ * s / (float)B_);
}

__global__ void finalK(const float* acc, float* outTotal) {
    float kl_loss = acc[0] / (float)(M_ * 255 * L_);
    outTotal[0] = acc[1] + kl_loss + acc[2];    // BETA = 1
}

// ---------------- host launch ----------------
extern "C" void kernel_launch(void* const* d_in, const int* in_sizes, int n_in,
                              void* d_out, int out_size, void* d_ws, size_t ws_size,
                              hipStream_t stream) {
    (void)in_sizes; (void)n_in; (void)out_size; (void)ws_size;
    const float* xs   = (const float*)d_in[0];
    const float* tree = (const float*)d_in[1];

    float* out    = (float*)d_out;
    float* outIdx = out;                                         // (M,DEPTH,B)
    float* outZ   = out + (size_t)M_ * DEPTH_ * B_;              // (M,DEPTH,B,L)
    float* outTot = outZ + (size_t)M_ * DEPTH_ * B_ * L_;        // scalar

    char* w = (char*)d_ws;
    float* acc    = (float*)w; w += 256;
    u16t*  xbf    = (u16t*)w;  w += (size_t)M_ * B_ * L_ * 2;           // 16 MB
    float* x2     = (float*)w; w += (size_t)M_ * B_ * 4;
    u16t*  ebf    = (u16t*)w;  w += (size_t)M_ * NODES_PAD * L_ * 2;
    u16t*  nbf    = (u16t*)w;  w += (size_t)M_ * NODES_PAD * L_ * 2;
    float* e2     = (float*)w; w += (size_t)M_ * NODES_PAD * 4;
    float* dot    = (float*)w; w += (size_t)M_ * B_ * NODES_PAD * 4;    // 32 MB
    float* gram   = (float*)w; w += (size_t)3 * NODES_PAD * NODES_PAD * 4;
    int*   cnt    = (int*)w;   w += (size_t)M_ * NODES_PAD * 4;
    u64t*  colmin = (u64t*)w;  w += (size_t)M_ * NODES_PAD * 8;
    float* lse    = (float*)w; w += (size_t)3 * NODES_PAD * 4;
    int*   idxi   = (int*)w;   w += (size_t)M_ * DEPTH_ * B_ * 4;

    initK<<<1, 1024, 0, stream>>>(acc, cnt, colmin);
    prepX<<<M_ * B_, 256, 0, stream>>>(xs, xbf, x2);
    prepE<<<M_ * NODES_PAD, 256, 0, stream>>>(tree, ebf, nbf, e2, acc);
    dotGemm<<<4096, 256, 0, stream>>>(xbf, ebf, dot);        // 32768 WMMA tiles
    gramGemm<<<96, 256, 0, stream>>>(nbf, gram);             // 768 WMMA tiles
    for (int d = 0; d < DEPTH_; ++d)
        levelK<<<(M_ * B_) / 256, 256, 0, stream>>>(x2, e2, dot, idxi, outIdx,
                                                    cnt, colmin, acc, d);
    colK<<<M_, 256, 0, stream>>>(x2, e2, dot, colmin, acc);
    zGather<<<(M_ * DEPTH_ * B_) / 8, 256, 0, stream>>>(tree, idxi, outZ);
    wassK<<<24, 128, 0, stream>>>(cnt, acc);
    lseK<<<24, 128, 0, stream>>>(gram, cnt, lse);
    nceK<<<(3 * DEPTH_ * B_) / 256, 256, 0, stream>>>(gram, lse, idxi, acc);
    finalK<<<1, 1, 0, stream>>>(acc, outTot);
}
